// Encoder_43009802502573
// MI455X (gfx1250) — matrix operand order repaired and checked by reading, not compiled, not run on hardware
//
#include <hip/hip_runtime.h>
#include <hip/hip_bf16.h>
#include <stdint.h>

#define B_  64
#define N0  2619
#define N1  655
#define N2  164
#define CH  64
#define SP  9
#define KP1 2624   // 2619 padded to mult of 32
#define KP2 672    // 655 padded to mult of 32
#define EPSF 1e-5f
#define CDIV(a,b) (((a)+(b)-1)/(b))

typedef __bf16 bf16;
typedef __attribute__((ext_vector_type(16))) __bf16 v16bf;
typedef __attribute__((ext_vector_type(8)))  __bf16 v8bf;
typedef __attribute__((ext_vector_type(8)))  float  v8f;

__device__ __forceinline__ bf16 bf0() {
    return __builtin_bit_cast(bf16, (unsigned short)0);
}
__device__ __forceinline__ bf16 f2bf(float f) {   // round-to-nearest-even truncation
    unsigned u = __builtin_bit_cast(unsigned, f);
    unsigned r = u + 0x7FFFu + ((u >> 16) & 1u);
    return __builtin_bit_cast(bf16, (unsigned short)(r >> 16));
}
__device__ __forceinline__ v8f wmma_bf16(v16bf a, v16bf b, v8f c) {
    return __builtin_amdgcn_wmma_f32_16x16x32_bf16(false, a, false, b, (short)0, c,
                                                   false, false);
}

// ---------------- prep kernels ----------------

// h = concat(vertices, x) -> bf16 [B, N0, 4]
__global__ void concat4_kernel(const float* __restrict__ x, const float* __restrict__ v,
                               bf16* __restrict__ out, int bn) {
    int i = blockIdx.x * 256 + threadIdx.x;
    if (i >= bn * 4) return;
    int c = i & 3, r = i >> 2;
    float val = (c < 3) ? v[(size_t)r * 3 + c] : x[r];
    out[i] = f2bf(val);
}

// fp32 [rows, kin] -> bf16 [rows, kpad], zero padded along K
__global__ void convert_pad_kernel(const float* __restrict__ src, bf16* __restrict__ dst,
                                   int rows, int kin, int kpad) {
    int i = blockIdx.x * 256 + threadIdx.x;
    if (i >= rows * kpad) return;
    int r = i / kpad, k = i - r * kpad;
    dst[i] = (k < kin) ? f2bf(src[(size_t)r * kin + k]) : bf0();
}

// ---------------- WMMA GEMM kernels ----------------
// Common mapping (wave32): lo = lane&15, hi = lane>>4.
// A frag elem e   : K = kb + 16*hi + e  ; row = mbase + lo
// B frag elem e   : K = kb + 16*hi + e  ; column = nbase + lo
// C frag elem r   : row = mbase + r + 8*hi ; col = nbase + lo
// Block = 128 threads = 4 waves, wave w owns columns [16w, 16w+16).

// Spiral block with C_in = 64: out[row, n] = sum_k gather(act)[row, k] * W[n, k]
// gather(act)[b*Nn+node, s*64+c] = act[b, idx[node,s], c];  K = 576
__global__ void gemm_spiral_c64(const bf16* __restrict__ act, const int* __restrict__ idx,
                                const bf16* __restrict__ W, float* __restrict__ out,
                                int Nn) {
    const int lane = threadIdx.x & 31;
    const int wv   = threadIdx.x >> 5;
    const int hi = lane >> 4, lo = lane & 15;
    const int row  = blockIdx.x * 16 + lo;
    const int b    = row / Nn;
    const int node = row - b * Nn;
    const int ncol = wv * 16 + lo;
    v8f acc = {};
    for (int kb = 0; kb < SP * CH; kb += 32) {
        const int s  = kb >> 6;         // both 8-chunks stay within one spiral slot
        const int c0 = kb & 63;
        const int src = idx[node * SP + s];
        const bf16* ap = act + ((size_t)b * Nn + src) * CH + c0;
        v8bf x0 = *(const v8bf*)(ap + 16 * hi);
        v8bf x1 = *(const v8bf*)(ap + 16 * hi + 8);
        v16bf a;
#pragma unroll
        for (int e = 0; e < 8; ++e) { a[e] = x0[e]; a[8 + e] = x1[e]; }
        v16bf bb = *(const v16bf*)(W + (size_t)ncol * (SP * CH) + kb + 16 * hi);
        acc = wmma_bf16(a, bb, acc);
    }
    const int mb = blockIdx.x * 16;
#pragma unroll
    for (int r = 0; r < 8; ++r)
        out[(size_t)(mb + r + 8 * hi) * CH + ncol] = acc[r];
}

// Spiral block 0 with C_in = 4 (K logical 36, padded to 64; W pre-padded)
__global__ void gemm_spiral_c4(const bf16* __restrict__ act4, const int* __restrict__ idx,
                               const bf16* __restrict__ W, float* __restrict__ out,
                               int Nn) {
    const int lane = threadIdx.x & 31;
    const int wv   = threadIdx.x >> 5;
    const int hi = lane >> 4, lo = lane & 15;
    const int row  = blockIdx.x * 16 + lo;
    const int b    = row / Nn;
    const int node = row - b * Nn;
    const int ncol = wv * 16 + lo;
    v8f acc = {};
    for (int kb = 0; kb < 64; kb += 32) {
        v16bf a;
#pragma unroll
        for (int e = 0; e < 16; ++e) {
            int k = kb + e + 16 * hi;
            bf16 val = bf0();
            if (k < 36) {
                int s = k >> 2, c = k & 3;
                val = act4[((size_t)b * Nn + idx[node * SP + s]) * 4 + c];
            }
            a[e] = val;
        }
        v16bf bb = *(const v16bf*)(W + (size_t)ncol * 64 + kb + 16 * hi);
        acc = wmma_bf16(a, bb, acc);
    }
    const int mb = blockIdx.x * 16;
#pragma unroll
    for (int r = 0; r < 8; ++r)
        out[(size_t)(mb + r + 8 * hi) * CH + ncol] = acc[r];
}

// Batched downsample: out[b,m,c] = sum_k Dw[m,k] * actT[b,c,k]
// Dw bf16 [M, Kpad] zero-padded; actT bf16 [B, 64, Kpad] (transposed, tail-zeroed)
// Writes bf16 directly (no norm follows).  grid = (ceil(M/16), B)
__global__ void gemm_down(const bf16* __restrict__ Dw, const bf16* __restrict__ actT,
                          bf16* __restrict__ out, int M, int Kpad) {
    const int lane = threadIdx.x & 31;
    const int wv   = threadIdx.x >> 5;
    const int hi = lane >> 4, lo = lane & 15;
    const int batch = blockIdx.y;
    const int m     = blockIdx.x * 16 + lo;
    const int ncol  = wv * 16 + lo;
    const bf16* bt = actT + ((size_t)batch * CH + ncol) * Kpad;
    v8f acc = {};
    for (int kb = 0; kb < Kpad; kb += 32) {
        v16bf a;
        if (m < M) {
            const bf16* ap = Dw + (size_t)m * Kpad + kb;
            v8bf x0 = *(const v8bf*)(ap + 16 * hi);
            v8bf x1 = *(const v8bf*)(ap + 16 * hi + 8);
#pragma unroll
            for (int e = 0; e < 8; ++e) { a[e] = x0[e]; a[8 + e] = x1[e]; }
        } else {
#pragma unroll
            for (int e = 0; e < 16; ++e) a[e] = bf0();
        }
        v16bf bb = *(const v16bf*)(bt + kb + 16 * hi);   // contiguous 32B, pad-safe
        acc = wmma_bf16(a, bb, acc);
    }
#pragma unroll
    for (int r = 0; r < 8; ++r) {
        int mm = blockIdx.x * 16 + r + 8 * hi;
        if (mm < M) out[((size_t)batch * M + mm) * CH + ncol] = f2bf(acc[r]);
    }
}

// Dense: out[m,n] = sum_k A[m,k] * W[n,k].  grid = (M/16, N/64)
__global__ void gemm_dense(const bf16* __restrict__ A, const bf16* __restrict__ W,
                           float* __restrict__ out, int K, int N) {
    const int lane = threadIdx.x & 31;
    const int wv   = threadIdx.x >> 5;
    const int hi = lane >> 4, lo = lane & 15;
    const int row  = blockIdx.x * 16 + lo;
    const int ncol = blockIdx.y * 64 + wv * 16 + lo;
    v8f acc = {};
    for (int kb = 0; kb < K; kb += 32) {
        const bf16* ap = A + (size_t)row * K + kb;
        v8bf x0 = *(const v8bf*)(ap + 16 * hi);
        v8bf x1 = *(const v8bf*)(ap + 16 * hi + 8);
        v16bf a;
#pragma unroll
        for (int e = 0; e < 8; ++e) { a[e] = x0[e]; a[8 + e] = x1[e]; }
        v16bf bb = *(const v16bf*)(W + (size_t)ncol * K + kb + 16 * hi);
        acc = wmma_bf16(a, bb, acc);
    }
    const int mb = blockIdx.x * 16;
#pragma unroll
    for (int r = 0; r < 8; ++r)
        out[(size_t)(mb + r + 8 * hi) * N + ncol] = acc[r];
}

// ---------------- normalization epilogues ----------------

// GroupNorm(8 groups over [nodes x 8ch]) + affine + ReLU -> bf16 [B, Nn, 64]
// grid = B*8, block = 256
__global__ void groupnorm_relu(const float* __restrict__ gin, const float* __restrict__ bias,
                               const float* __restrict__ gamma, const float* __restrict__ beta,
                               bf16* __restrict__ out, int Nn) {
    const int b = blockIdx.x >> 3;
    const int g = blockIdx.x & 7;
    const int tid = threadIdx.x;
    const int total = Nn * 8;
    __shared__ float ss[256], sq[256];
    float s = 0.f, q = 0.f;
    for (int t = tid; t < total; t += 256) {
        int node = t >> 3, c = g * 8 + (t & 7);
        float x = gin[((size_t)b * Nn + node) * CH + c] + bias[c];
        s += x; q += x * x;
    }
    ss[tid] = s; sq[tid] = q; __syncthreads();
    for (int off = 128; off > 0; off >>= 1) {
        if (tid < off) { ss[tid] += ss[tid + off]; sq[tid] += sq[tid + off]; }
        __syncthreads();
    }
    float inv  = 1.f / (float)total;
    float mean = ss[0] * inv;
    float var  = sq[0] * inv - mean * mean;   // biased, matches jnp.var
    float rstd = rsqrtf(var + EPSF);
    for (int t = tid; t < total; t += 256) {
        int node = t >> 3, c = g * 8 + (t & 7);
        size_t o = ((size_t)b * Nn + node) * CH + c;
        float x = gin[o] + bias[c];
        float y = (x - mean) * rstd * gamma[c] + beta[c];
        out[o] = f2bf(fmaxf(y, 0.f));
    }
}

// Same, but writes TRANSPOSED bf16 [B, 64, Kpad] with zeroed K tail — feeds gemm_down
// so the downsample B-frag becomes a contiguous 32B load.  grid = B*8, block = 256
__global__ void groupnorm_relu_T(const float* __restrict__ gin, const float* __restrict__ bias,
                                 const float* __restrict__ gamma, const float* __restrict__ beta,
                                 bf16* __restrict__ outT, int Nn, int Kpad) {
    const int b = blockIdx.x >> 3;
    const int g = blockIdx.x & 7;
    const int tid = threadIdx.x;
    const int total = Nn * 8;
    __shared__ float ss[256], sq[256];
    float s = 0.f, q = 0.f;
    for (int t = tid; t < total; t += 256) {
        int node = t >> 3, c = g * 8 + (t & 7);
        float x = gin[((size_t)b * Nn + node) * CH + c] + bias[c];
        s += x; q += x * x;
    }
    ss[tid] = s; sq[tid] = q; __syncthreads();
    for (int off = 128; off > 0; off >>= 1) {
        if (tid < off) { ss[tid] += ss[tid + off]; sq[tid] += sq[tid + off]; }
        __syncthreads();
    }
    float inv  = 1.f / (float)total;
    float mean = ss[0] * inv;
    float var  = sq[0] * inv - mean * mean;
    float rstd = rsqrtf(var + EPSF);
    for (int t = tid; t < total; t += 256) {
        int node = t >> 3, c = g * 8 + (t & 7);
        float x = gin[((size_t)b * Nn + node) * CH + c] + bias[c];
        float y = (x - mean) * rstd * gamma[c] + beta[c];
        outT[((size_t)b * CH + c) * Kpad + node] = f2bf(fmaxf(y, 0.f));
    }
    // zero the padded K tail for this (b, g)'s 8 channels
    const int padn = Kpad - Nn;
    for (int t = tid; t < padn * 8; t += 256) {
        int node = Nn + (t >> 3), c = g * 8 + (t & 7);
        outT[((size_t)b * CH + c) * Kpad + node] = bf0();
    }
}

// LayerNorm over 512 + affine + ReLU -> bf16.  grid = B, block = 256
__global__ void layernorm_relu(const float* __restrict__ gin, const float* __restrict__ bias,
                               const float* __restrict__ gamma, const float* __restrict__ beta,
                               bf16* __restrict__ out) {
    const int row = blockIdx.x, tid = threadIdx.x;
    __shared__ float ss[256], sq[256];
    float s = 0.f, q = 0.f;
    for (int c = tid; c < 512; c += 256) {
        float x = gin[(size_t)row * 512 + c] + bias[c];
        s += x; q += x * x;
    }
    ss[tid] = s; sq[tid] = q; __syncthreads();
    for (int off = 128; off > 0; off >>= 1) {
        if (tid < off) { ss[tid] += ss[tid + off]; sq[tid] += sq[tid + off]; }
        __syncthreads();
    }
    float mean = ss[0] * (1.f / 512.f);
    float var  = sq[0] * (1.f / 512.f) - mean * mean;
    float rstd = rsqrtf(var + EPSF);
    for (int c = tid; c < 512; c += 256) {
        float x = gin[(size_t)row * 512 + c] + bias[c];
        float y = (x - mean) * rstd * gamma[c] + beta[c];
        out[(size_t)row * 512 + c] = f2bf(fmaxf(y, 0.f));
    }
}

// d_out = [mu (64x256) ; log_var (64x256)] with biases
__global__ void bias_out_kernel(const float* __restrict__ muf, const float* __restrict__ lvf,
                                const float* __restrict__ bmu, const float* __restrict__ blv,
                                float* __restrict__ out) {
    int i = blockIdx.x * 256 + threadIdx.x;
    if (i < B_ * 256)            out[i] = muf[i] + bmu[i & 255];
    else if (i < 2 * B_ * 256)   out[i] = lvf[i - B_ * 256] + blv[i & 255];
}

// ---------------- host orchestration ----------------

extern "C" void kernel_launch(void* const* d_in, const int* in_sizes, int n_in,
                              void* d_out, int out_size, void* d_ws, size_t ws_size,
                              hipStream_t stream) {
    (void)in_sizes; (void)n_in; (void)out_size; (void)ws_size;
    const float* x    = (const float*)d_in[0];
    const float* vert = (const float*)d_in[1];
    const int*   idx0 = (const int*)d_in[2];
    const int*   idx1 = (const int*)d_in[3];
    const int*   idx2 = (const int*)d_in[4];
    const float* D1   = (const float*)d_in[5];
    const float* D2   = (const float*)d_in[6];
    const float* W0 = (const float*)d_in[7],  *b0 = (const float*)d_in[8];
    const float* g0 = (const float*)d_in[9],  *be0 = (const float*)d_in[10];
    const float* W1 = (const float*)d_in[11], *b1 = (const float*)d_in[12];
    const float* g1 = (const float*)d_in[13], *be1 = (const float*)d_in[14];
    const float* W2 = (const float*)d_in[15], *b2 = (const float*)d_in[16];
    const float* g2 = (const float*)d_in[17], *be2 = (const float*)d_in[18];
    const float* W3 = (const float*)d_in[19], *b3 = (const float*)d_in[20];
    const float* be3 = (const float*)d_in[21], *g3 = (const float*)d_in[22]; // note dict order
    const float* Wfc = (const float*)d_in[23], *bfc = (const float*)d_in[24];
    const float* lng = (const float*)d_in[25], *lnb = (const float*)d_in[26];
    const float* Wmu = (const float*)d_in[27], *bmu = (const float*)d_in[28];
    const float* Wlv = (const float*)d_in[29], *blv = (const float*)d_in[30];

    char* p = (char*)d_ws;
    auto alloc = [&](size_t bytes) -> void* {
        void* r = (void*)p; p += (bytes + 255) & ~(size_t)255; return r;
    };
    bf16* a_in  = (bf16*)alloc((size_t)B_ * N0 * 4 * 2);
    bf16* actA  = (bf16*)alloc((size_t)B_ * N0 * CH * 2);
    // actB doubles as the transposed activation [B,64,KP1]; size = max of both uses
    bf16* actB  = (bf16*)alloc((size_t)B_ * CH * KP1 * 2);
    bf16* actFc = (bf16*)alloc((size_t)B_ * 512 * 2);
    bf16* w0p  = (bf16*)alloc((size_t)64 * 64 * 2);
    bf16* w1b  = (bf16*)alloc((size_t)64 * 576 * 2);
    bf16* w2b  = (bf16*)alloc((size_t)64 * 576 * 2);
    bf16* w3b  = (bf16*)alloc((size_t)64 * 576 * 2);
    bf16* d1b  = (bf16*)alloc((size_t)655 * KP1 * 2);
    bf16* d2b  = (bf16*)alloc((size_t)164 * KP2 * 2);
    bf16* wfcb = (bf16*)alloc((size_t)512 * 10496 * 2);
    bf16* wmub = (bf16*)alloc((size_t)256 * 512 * 2);
    bf16* wlvb = (bf16*)alloc((size_t)256 * 512 * 2);
    float* gemmf = (float*)alloc((size_t)B_ * N0 * CH * 4);
    float* fcf   = (float*)alloc((size_t)B_ * 512 * 4);
    float* muf   = (float*)alloc((size_t)B_ * 256 * 4);
    float* lvf   = (float*)alloc((size_t)B_ * 256 * 4);

    // input + weight conversion (bf16, zero-padded K)
    concat4_kernel<<<CDIV(B_ * N0 * 4, 256), 256, 0, stream>>>(x, vert, a_in, B_ * N0);
    convert_pad_kernel<<<CDIV(64 * 64, 256), 256, 0, stream>>>(W0, w0p, 64, 36, 64);
    convert_pad_kernel<<<CDIV(64 * 576, 256), 256, 0, stream>>>(W1, w1b, 64, 576, 576);
    convert_pad_kernel<<<CDIV(64 * 576, 256), 256, 0, stream>>>(W2, w2b, 64, 576, 576);
    convert_pad_kernel<<<CDIV(64 * 576, 256), 256, 0, stream>>>(W3, w3b, 64, 576, 576);
    convert_pad_kernel<<<CDIV(655 * KP1, 256), 256, 0, stream>>>(D1, d1b, 655, 2619, KP1);
    convert_pad_kernel<<<CDIV(164 * KP2, 256), 256, 0, stream>>>(D2, d2b, 164, 655, KP2);
    convert_pad_kernel<<<CDIV(512 * 10496, 256), 256, 0, stream>>>(Wfc, wfcb, 512, 10496, 10496);
    convert_pad_kernel<<<CDIV(256 * 512, 256), 256, 0, stream>>>(Wmu, wmub, 256, 512, 512);
    convert_pad_kernel<<<CDIV(256 * 512, 256), 256, 0, stream>>>(Wlv, wlvb, 256, 512, 512);

    // block 0
    gemm_spiral_c4<<<B_ * N0 / 16, 128, 0, stream>>>(a_in, idx0, w0p, gemmf, N0);
    groupnorm_relu<<<B_ * 8, 256, 0, stream>>>(gemmf, b0, g0, be0, actA, N0);
    // block 1 (GN writes transposed: only consumer is downsample 1)
    gemm_spiral_c64<<<B_ * N0 / 16, 128, 0, stream>>>(actA, idx0, w1b, gemmf, N0);
    groupnorm_relu_T<<<B_ * 8, 256, 0, stream>>>(gemmf, b1, g1, be1, actB, N0, KP1);
    // downsample 1: 2619 -> 655, writes bf16 directly
    gemm_down<<<dim3(CDIV(655, 16), B_), 128, 0, stream>>>(d1b, actB, actA, 655, KP1);
    // block 2 (GN transposed: only consumer is downsample 2)
    gemm_spiral_c64<<<B_ * N1 / 16, 128, 0, stream>>>(actA, idx1, w2b, gemmf, N1);
    groupnorm_relu_T<<<B_ * 8, 256, 0, stream>>>(gemmf, b2, g2, be2, actB, N1, KP2);
    // downsample 2: 655 -> 164
    gemm_down<<<dim3(CDIV(164, 16), B_), 128, 0, stream>>>(d2b, actB, actA, 164, KP2);
    // block 3 (normal layout: feeds the flatten->FC)
    gemm_spiral_c64<<<B_ * N2 / 16, 128, 0, stream>>>(actA, idx2, w3b, gemmf, N2);
    groupnorm_relu<<<B_ * 8, 256, 0, stream>>>(gemmf, b3, g3, be3, actB, N2);
    // FC 10496 -> 512, LayerNorm + ReLU
    gemm_dense<<<dim3(B_ / 16, 512 / 64), 128, 0, stream>>>(actB, wfcb, fcf, N2 * CH, 512);
    layernorm_relu<<<B_, 256, 0, stream>>>(fcf, bfc, lng, lnb, actFc);
    // heads
    gemm_dense<<<dim3(B_ / 16, 256 / 64), 128, 0, stream>>>(actFc, wmub, muf, 512, 256);
    gemm_dense<<<dim3(B_ / 16, 256 / 64), 128, 0, stream>>>(actFc, wlvb, lvf, 512, 256);
    bias_out_kernel<<<CDIV(2 * B_ * 256, 256), 256, 0, stream>>>(muf, lvf, bmu, blv,
                                                                 (float*)d_out);
}
